// GeometricDynamicsPredictor_54511724921355
// MI455X (gfx1250) — matrix operand order repaired and checked by reading, not compiled, not run on hardware
//
#include <hip/hip_runtime.h>
#include <hip/hip_bf16.h>

// ---------------------------------------------------------------------------
// CDNA5 (gfx1250) Kuramoto RK4 — bf16 WMMA + TDM double-buffered pipeline.
// Workspace requirement: 8*2097152 f32 + 2*2097152 bf16 = ~72 MB.
// ---------------------------------------------------------------------------

typedef __attribute__((ext_vector_type(16))) __bf16   v16bf;
typedef __attribute__((ext_vector_type(8)))  __bf16   v8bf;
typedef __attribute__((ext_vector_type(8)))  float    v8f;
typedef __attribute__((ext_vector_type(4)))  uint32_t u32x4;
typedef __attribute__((ext_vector_type(8)))  uint32_t u32x8;

__device__ __forceinline__ v16bf cat8(v8bf a, v8bf b) {
    return __builtin_shufflevector(a, b, 0,1,2,3,4,5,6,7,8,9,10,11,12,13,14,15);
}
__device__ __forceinline__ v8bf cvt8(v8f x) {
    v8bf r;
#pragma unroll
    for (int i = 0; i < 8; ++i) r[i] = (__bf16)x[i];
    return r;
}

// ---------------------------------------------------------------------------
// TDM: DMA a contiguous 32x64 bf16 tile (4 KB) from global into LDS.
// Descriptor per CDNA5 ISA §8.3/8.4:
//   group0: count=1 | lds_addr | global_addr[56:0] | type=2
//   group1: data_size=1 (2B), tensor_dim0=64, tensor_dim1=32,
//           tile_dim0=64, tile_dim1=32, tensor_dim0_stride=64,
//           tensor_dim1_stride=2048
// Issued once per wave (EXEC ignored); tracked with TENSORcnt.
// ---------------------------------------------------------------------------
__device__ __forceinline__ void tdm_load_tile_32x64_bf16(uint32_t lds_byte,
                                                         uint64_t gaddr) {
    u32x4 g0;
    g0[0] = 1u;                                              // count=1 (valid)
    g0[1] = lds_byte;                                        // lds_addr
    g0[2] = (uint32_t)gaddr;                                 // global_addr lo
    g0[3] = ((uint32_t)(gaddr >> 32) & 0x01FFFFFFu)          // global_addr hi
            | 0x80000000u;                                   // type=2 (image)
    u32x8 g1;
    g1[0] = 0x00010000u;          // workgroup_mask=0, data_size=1 (2 bytes)
    g1[1] = 64u << 16;            // tensor_dim0[15:0] = 64
    g1[2] = 32u << 16;            // tensor_dim1[15:0] = 32
    g1[3] = 64u << 16;            // tile_dim0 = 64
    g1[4] = 32u;                  // tile_dim1 = 32, tile_dim2 = 0
    g1[5] = 64u;                  // tensor_dim0_stride = 64 (lo 32)
    g1[6] = 2048u << 16;          // stride0 hi=0, tensor_dim1_stride lo16
    g1[7] = 0u;                   // tensor_dim1_stride hi
    asm volatile("tensor_load_to_lds %0, %1" :: "s"(g0), "s"(g1) : "memory");
}

// ---------------------------------------------------------------------------
// GEMM: C[M,N] = A[M,K] * W[N,K]^T  (+ bias1 + bias2), f32 in/out, bf16 WMMA.
// One wave per block computes a 32x64 tile: 2 A-fragments x 4 B-subtiles
// -> 8 WMMA per 32-wide K step (B fragments reused across both A frags).
// ---------------------------------------------------------------------------
__global__ __launch_bounds__(32) void wmma_gemm_nt(
    const float* __restrict__ A, const float* __restrict__ W,
    float* __restrict__ C, int N, int K,
    const float* __restrict__ bias1, const float* __restrict__ bias2)
{
    const int mt = blockIdx.x * 32;
    const int nt = blockIdx.y * 64;
    const int L  = threadIdx.x;
    const int g  = L >> 4;
    const int ln = L & 15;

    v8f c[2][4] = {};
    const float* arow0 = A + (size_t)(mt + ln) * K;
    const float* arow1 = A + (size_t)(mt + 16 + ln) * K;

    for (int kb = 0; kb < K; kb += 32) {
        const int kg = kb + g * 8;
        if (kb + 32 < K) __builtin_prefetch(arow0 + kb + 32, 0, 1);
        v16bf a0 = cat8(cvt8(*(const v8f*)(arow0 + kg)),
                        cvt8(*(const v8f*)(arow0 + kg + 16)));
        v16bf a1 = cat8(cvt8(*(const v8f*)(arow1 + kg)),
                        cvt8(*(const v8f*)(arow1 + kg + 16)));
#pragma unroll
        for (int t = 0; t < 4; ++t) {
            const float* wp = W + (size_t)(nt + t * 16 + ln) * K + kb + g * 8;
            v16bf bf = cat8(cvt8(*(const v8f*)wp), cvt8(*(const v8f*)(wp + 16)));
            c[0][t] = __builtin_amdgcn_wmma_f32_16x16x32_bf16(
                false, a0, false, bf, (short)0, c[0][t], false, false);
            c[1][t] = __builtin_amdgcn_wmma_f32_16x16x32_bf16(
                false, a1, false, bf, (short)0, c[1][t], false, false);
        }
    }

#pragma unroll
    for (int a = 0; a < 2; ++a)
#pragma unroll
        for (int t = 0; t < 4; ++t) {
            const int n = nt + t * 16 + ln;
            float add = 0.f;
            if (bias1) add += bias1[n];
            if (bias2) add += bias2[n];
#pragma unroll
            for (int r = 0; r < 8; ++r)
                C[(size_t)(mt + a * 16 + r + 8 * g) * N + n] = c[a][t][r] + add;
        }
}

// ---------------------------------------------------------------------------
// Per-head L2 normalize of q|k (qk: [B*T, 512]) -> bf16 in [B,H,T,64] layout.
// ---------------------------------------------------------------------------
__global__ __launch_bounds__(256) void head_norm_kernel(
    const float* __restrict__ qk, __bf16* __restrict__ qn,
    __bf16* __restrict__ kn, int T)
{
    const int row = blockIdx.x;          // b*T + t
    const int d   = threadIdx.x;         // 0..255
    const int b   = row / T, t = row % T;

    const float qv = qk[(size_t)row * 512 + d];
    const float kv = qk[(size_t)row * 512 + 256 + d];

    float q2 = qv * qv, k2 = kv * kv;
#pragma unroll
    for (int off = 16; off; off >>= 1) {
        q2 += __shfl_xor(q2, off, 32);
        k2 += __shfl_xor(k2, off, 32);
    }
    __shared__ float sq[8], sk[8];
    if ((d & 31) == 0) { sq[d >> 5] = q2; sk[d >> 5] = k2; }
    __syncthreads();

    const int h = d >> 6;
    const float qinv = 1.f / fmaxf(sqrtf(sq[2 * h] + sq[2 * h + 1]), 1e-12f);
    const float kinv = 1.f / fmaxf(sqrtf(sk[2 * h] + sk[2 * h + 1]), 1e-12f);

    const size_t o = ((size_t)(b * 4 + h) * T + t) * 64 + (d & 63);
    qn[o] = (__bf16)(qv * qinv);
    kn[o] = (__bf16)(kv * kinv);
}

// ---------------------------------------------------------------------------
// Attention-like kernel: F[q,d] = sum_k sin(q . k) * k   per (b,h).
// Block = 128 threads (4 waves); each wave owns 16 q-rows.
// k-tiles (32x64 bf16) are TDM-DMA'd into double-buffered LDS; the load for
// tile i+1 overlaps compute on tile i (wait with s_wait_tensorcnt <= 1).
// ---------------------------------------------------------------------------
__global__ __launch_bounds__(128) void attn_kernel(
    const __bf16* __restrict__ qn, const __bf16* __restrict__ kn,
    float* __restrict__ force, int T)
{
    __shared__ __bf16 kTile[2][32 * 64]; // [buf][k'][d] — TDM destination
    __shared__ __bf16 kTileT[64 * 32];   // [d][k']
    __shared__ __bf16 sTile[4 * 16 * 32];// per-wave sin(S) staging

    const int bh = blockIdx.y;
    const int b = bh >> 2, h = bh & 3;
    const __bf16* qb = qn + (size_t)bh * T * 64;
    const __bf16* kb = kn + (size_t)bh * T * 64;

    const int tid = threadIdx.x;
    const int wave = tid >> 5, L = tid & 31, g = L >> 4, ln = L & 15;
    const int q0 = blockIdx.x * 64 + wave * 16;

    // Q fragments for this wave's 16 rows: K = d split into two 16x32 frags.
    v16bf aq0, aq1;
    {
        const __bf16* qp = qb + (size_t)(q0 + ln) * 64;
        aq0 = cat8(*(const v8bf*)(qp + g * 8),      *(const v8bf*)(qp + g * 8 + 16));
        aq1 = cat8(*(const v8bf*)(qp + 32 + g * 8), *(const v8bf*)(qp + 32 + g * 8 + 16));
    }

    // Prologue: TDM-load tile 0 into buffer 0 (one instruction, wave 0 only).
    if (wave == 0)
        tdm_load_tile_32x64_bf16((uint32_t)(uintptr_t)&kTile[0][0],
                                 (uint64_t)(uintptr_t)kb);

    v8f F[4] = {};

    for (int kt = 0; kt < T; kt += 32) {
        const int  cur  = (kt >> 5) & 1;
        const bool more = (kt + 32) < T;

        if (wave == 0) {
            if (more)
                tdm_load_tile_32x64_bf16(
                    (uint32_t)(uintptr_t)&kTile[cur ^ 1][0],
                    (uint64_t)(uintptr_t)(kb + (size_t)(kt + 32) * 64));
            if (more) __builtin_amdgcn_s_wait_tensorcnt((short)1); // tile kt done
            else      __builtin_amdgcn_s_wait_tensorcnt((short)0);
        }
        __syncthreads();                 // tile kt visible to all waves

        const __bf16* kc = kTile[cur];

        {   // build transposed copy [d][k']
            const int kp = tid & 31, d0 = (tid >> 5) * 16;
#pragma unroll
            for (int j = 0; j < 16; ++j)
                kTileT[(d0 + j) * 32 + kp] = kc[kp * 64 + d0 + j];
        }
        __syncthreads();

        // ---- S = q . k^T : two 16x16 C tiles (k' cols 0-15 / 16-31), K=64 ----
        v8f s0 = {}, s1 = {};
        {
            const __bf16* p0 = kc + ln * 64 + g * 8;
            const __bf16* p1 = kc + (ln + 16) * 64 + g * 8;
            v16bf b0 = cat8(*(const v8bf*)p0, *(const v8bf*)(p0 + 16));
            v16bf b1 = cat8(*(const v8bf*)p1, *(const v8bf*)(p1 + 16));
            s0 = __builtin_amdgcn_wmma_f32_16x16x32_bf16(false, aq0, false, b0, (short)0, s0, false, false);
            s1 = __builtin_amdgcn_wmma_f32_16x16x32_bf16(false, aq0, false, b1, (short)0, s1, false, false);
            b0 = cat8(*(const v8bf*)(p0 + 32), *(const v8bf*)(p0 + 48));
            b1 = cat8(*(const v8bf*)(p1 + 32), *(const v8bf*)(p1 + 48));
            s0 = __builtin_amdgcn_wmma_f32_16x16x32_bf16(false, aq1, false, b0, (short)0, s0, false, false);
            s1 = __builtin_amdgcn_wmma_f32_16x16x32_bf16(false, aq1, false, b1, (short)0, s1, false, false);
        }

        // ---- sin + re-layout C->A via per-wave LDS staging ----
        __bf16* st = sTile + wave * 512;
#pragma unroll
        for (int r = 0; r < 8; ++r) {
            st[(r + 8 * g) * 32 + ln]      = (__bf16)__sinf(s0[r]);
            st[(r + 8 * g) * 32 + 16 + ln] = (__bf16)__sinf(s1[r]);
        }
        // wave-private region; LDS is in-order per wave — fence + wait
        asm volatile("s_wait_dscnt 0" ::: "memory");

        v16bf as;
        {
            const __bf16* sp = st + ln * 32 + g * 8;
            as = cat8(*(const v8bf*)sp, *(const v8bf*)(sp + 16));
        }

        // ---- F += sin(S) . k : 4 N-subtiles over d, K = k' = 32 ----
#pragma unroll
        for (int t = 0; t < 4; ++t) {
            const __bf16* bp = kTileT + (t * 16 + ln) * 32 + g * 8;
            v16bf bk = cat8(*(const v8bf*)bp, *(const v8bf*)(bp + 16));
            F[t] = __builtin_amdgcn_wmma_f32_16x16x32_bf16(
                false, as, false, bk, (short)0, F[t], false, false);
        }
        __syncthreads();   // all reads of kTile[cur]/kTileT done before refill
    }

    float* ob = force + (size_t)b * T * 256 + h * 64;
#pragma unroll
    for (int t = 0; t < 4; ++t)
#pragma unroll
        for (int r = 0; r < 8; ++r)
            ob[(size_t)(q0 + r + 8 * g) * 256 + t * 16 + ln] = F[t][r];
}

// ---------------------------------------------------------------------------
// Elementwise helpers
// ---------------------------------------------------------------------------
__global__ __launch_bounds__(256) void l2norm_kernel(
    const float* __restrict__ in, float* __restrict__ out)
{
    const int row = blockIdx.x;
    const int d = threadIdx.x;
    const float v = in[(size_t)row * 256 + d];
    float s = v * v;
#pragma unroll
    for (int off = 16; off; off >>= 1) s += __shfl_xor(s, off, 32);
    __shared__ float sw[8];
    if ((d & 31) == 0) sw[d >> 5] = s;
    __syncthreads();
    float tot = 0.f;
#pragma unroll
    for (int i = 0; i < 8; ++i) tot += sw[i];
    out[(size_t)row * 256 + d] = v / fmaxf(sqrtf(tot), 1e-12f);
}

__global__ __launch_bounds__(256) void axpby_kernel(
    float* __restrict__ dst, const float* __restrict__ x,
    const float* __restrict__ y, float a, float bcoef, int n)
{
    const int i = blockIdx.x * blockDim.x + threadIdx.x;
    if (i < n) dst[i] = a * x[i] + bcoef * y[i];
}

__global__ __launch_bounds__(256) void acc_kernel(
    float* __restrict__ acc, const float* __restrict__ k,
    float w, int init, int n)
{
    const int i = blockIdx.x * blockDim.x + threadIdx.x;
    if (i < n) acc[i] = (init ? 0.f : acc[i]) + w * k[i];
}

// ---------------------------------------------------------------------------
// Driver
// ---------------------------------------------------------------------------
extern "C" void kernel_launch(void* const* d_in, const int* in_sizes, int n_in,
                              void* d_out, int out_size, void* d_ws, size_t ws_size,
                              hipStream_t stream) {
    (void)in_sizes; (void)n_in; (void)out_size; (void)ws_size;
    constexpr int B = 4, T = 2048, D = 256, H = 4;
    constexpr int M = B * T;                 // 8192 rows
    constexpr size_t NM = (size_t)M * D;     // 2097152 elements

    const float* z     = (const float*)d_in[0];
    const float* omega = (const float*)d_in[1];   // [D]
    const float* Wqk   = (const float*)d_in[2];   // [2D, D]
    const float* Wout  = (const float*)d_in[3];   // [D, D]
    const float* bout  = (const float*)d_in[4];   // [D]
    float* out = (float*)d_out;

    float* ws    = (float*)d_ws;
    float* xbuf  = ws;
    float* xn    = ws + 1 * NM;
    float* xt    = ws + 2 * NM;
    float* kcur  = ws + 3 * NM;
    float* acc   = ws + 4 * NM;
    float* qkbuf = ws + 5 * NM;              // 2*NM floats
    float* force = ws + 7 * NM;
    __bf16* qnb  = (__bf16*)(ws + 8 * NM);
    __bf16* knb  = qnb + NM;

    hipMemcpyAsync(xbuf, z, NM * sizeof(float), hipMemcpyDeviceToDevice, stream);

    const float dt = 0.25f;
    auto runF = [&](const float* xin, float* kout) {
        wmma_gemm_nt<<<dim3(M / 32, (2 * D) / 64), 32, 0, stream>>>(
            xin, Wqk, qkbuf, 2 * D, D, nullptr, nullptr);
        head_norm_kernel<<<dim3(M), 256, 0, stream>>>(qkbuf, qnb, knb, T);
        attn_kernel<<<dim3(T / 64, B * H), 128, 0, stream>>>(qnb, knb, force, T);
        wmma_gemm_nt<<<dim3(M / 32, D / 64), 32, 0, stream>>>(
            force, Wout, kout, D, D, bout, omega);
    };

    const int n = (int)NM;
    const int eb = 256, eg = (n + eb - 1) / eb;

    for (int s = 0; s < 4; ++s) {
        l2norm_kernel<<<dim3(M), 256, 0, stream>>>(xbuf, xn);

        runF(xn, kcur);                                                     // k1
        acc_kernel<<<eg, eb, 0, stream>>>(acc, kcur, 1.f, 1, n);
        axpby_kernel<<<eg, eb, 0, stream>>>(xt, xn, kcur, 1.f, dt * 0.5f, n);

        runF(xt, kcur);                                                     // k2
        acc_kernel<<<eg, eb, 0, stream>>>(acc, kcur, 2.f, 0, n);
        axpby_kernel<<<eg, eb, 0, stream>>>(xt, xn, kcur, 1.f, dt * 0.5f, n);

        runF(xt, kcur);                                                     // k3
        acc_kernel<<<eg, eb, 0, stream>>>(acc, kcur, 2.f, 0, n);
        axpby_kernel<<<eg, eb, 0, stream>>>(xt, xn, kcur, 1.f, dt, n);

        runF(xt, kcur);                                                     // k4
        acc_kernel<<<eg, eb, 0, stream>>>(acc, kcur, 1.f, 0, n);

        axpby_kernel<<<eg, eb, 0, stream>>>(xbuf, xn, acc, 1.f, dt / 6.f, n);
    }

    l2norm_kernel<<<dim3(M), 256, 0, stream>>>(xbuf, out);
}
